// VectorQuantizer_88510686036643
// MI455X (gfx1250) — matrix operand order repaired and checked by reading, not compiled, not run on hardware
//
#include <hip/hip_runtime.h>

// Problem constants (from reference): B=4, C=256, D=H=W=32, K=1024
#define Cc 256
#define Ss 32768          // D*H*W
#define Nn 131072         // B*S
#define Kk 1024
#define ZQ_ELEMS 33554432 // B*C*S

#define CODES_PER_STAGE 64          // 4 k-tiles of 16 codes per LDS stage
#define NSTAGES (Kk / CODES_PER_STAGE)

typedef __attribute__((ext_vector_type(16))) __bf16 v16bf;
typedef __attribute__((ext_vector_type(8)))  __bf16 v8bf;
typedef __attribute__((ext_vector_type(8)))  float  v8f;

__device__ __forceinline__ unsigned short f2bf(float f) {
  unsigned int u = __float_as_uint(f);
  if ((u & 0x7fffffffu) > 0x7f800000u) return (unsigned short)((u >> 16) | 0x0040u); // NaN
  u += 0x7fffu + ((u >> 16) & 1u); // round-to-nearest-even
  return (unsigned short)(u >> 16);
}

// ---- Kernel 1: codebook -> bf16 + per-code squared norms (fp32) ------------
__global__ void __launch_bounds__(256) vq_prep_emb(const float* __restrict__ emb,
                                                   unsigned short* __restrict__ ebuf,
                                                   float* __restrict__ enorm) {
  int k = blockIdx.x;
  int c = threadIdx.x;
  float v = emb[k * Cc + c];
  ebuf[k * Cc + c] = f2bf(v);
  float s = v * v;
  #pragma unroll
  for (int m = 16; m >= 1; m >>= 1) s += __shfl_xor(s, m, 32);
  __shared__ float red[8];
  if ((threadIdx.x & 31) == 0) red[threadIdx.x >> 5] = s;
  __syncthreads();
  if (threadIdx.x == 0) {
    float t = 0.f;
    #pragma unroll
    for (int i = 0; i < 8; ++i) t += red[i];
    enorm[k] = t;
  }
}

// ---- Kernel 2: fused distance GEMM (bf16 WMMA) + argmin over K -------------
// Block = 256 threads = 8 waves; block owns 128 consecutive n-rows, wave w owns
// rows [w*16, w*16+16).  zt holds bf16(-2*z) so dist = ||e||^2 + accumulator.
// Codebook is staged through LDS with double-buffered async copies (ASYNCcnt),
// 64 codes (32 KiB) per stage, so all 8 waves share one L2 read of each tile.
__global__ void __launch_bounds__(256) vq_argmin(const float* __restrict__ z,
                                                 const unsigned short* __restrict__ ebuf,
                                                 const float* __restrict__ enorm,
                                                 int* __restrict__ idx_out) {
  __shared__ __align__(16) unsigned short zt[128][264];                 // +8 pad
  __shared__ __align__(16) unsigned short et[2][CODES_PER_STAGE][264];  // 2 x 33 KiB

  const int tid = threadIdx.x;
  const int bid = blockIdx.x;

  // Async-copy one 64-code stage (64 rows x 512B) into et[buf]; row stride in
  // LDS is 528B (pad).  2048 x 16B transfers, 8 per thread, ASYNCcnt-tracked.
  auto stage_copy = [&](int stage, int buf) {
    const unsigned short* src = ebuf + (size_t)stage * CODES_PER_STAGE * Cc;
    #pragma unroll
    for (int j = 0; j < 8; ++j) {
      int flat = tid + (j << 8);          // 0..2047
      int code = flat >> 5;               // 32 x 16B chunks per code row
      int chnk = flat & 31;
      unsigned lds = (unsigned)(size_t)&et[buf][code][chnk * 8];
      unsigned long long ga = (unsigned long long)(size_t)(src + code * Cc + chnk * 8);
      asm volatile("global_load_async_to_lds_b128 %0, %1, off"
                   :: "v"(lds), "v"(ga) : "memory");
    }
  };

  stage_copy(0, 0);  // stage 0 in flight while we convert the z tile

  const int b  = (bid * 128) >> 15;       // / Ss
  const int s0 = (bid * 128) & (Ss - 1);
  const float* zbase = z + (size_t)b * Cc * Ss + s0;

  // Coalesced along n (innermost spatial dim of z), strided along c.
  #pragma unroll 4
  for (int i = 0; i < 128; ++i) {
    int flat = tid + (i << 8);
    int r = flat & 127, c = flat >> 7;
    zt[r][c] = f2bf(-2.0f * zbase[(size_t)c * Ss + r]);
  }
  __syncthreads();

  const int wave = tid >> 5;
  const int lane = tid & 31;
  const int mrow = (wave << 4) + (lane & 15);
  const int hi   = lane >> 4;   // which 16-lane half
  const int col  = lane & 15;   // output column within 16-wide k tile

  // A fragments (16-bit A 16x32 layout, ISA 7.12.2), resident in VGPRs.
  v16bf afrag[8];
  #pragma unroll
  for (int cs = 0; cs < 8; ++cs) {
    union { v8bf h[2]; v16bf v; } u;
    u.h[0] = *(const v8bf*)&zt[mrow][cs * 32 + hi * 8];
    u.h[1] = *(const v8bf*)&zt[mrow][cs * 32 + 16 + hi * 8];
    afrag[cs] = u.v;
  }

  float bestd[8]; int besti[8];
  #pragma unroll
  for (int r = 0; r < 8; ++r) { bestd[r] = 3.4028235e38f; besti[r] = 0; }

  for (int stage = 0; stage < NSTAGES; ++stage) {
    const int cur = stage & 1;
    asm volatile("s_wait_asynccnt 0x0" ::: "memory"); // et[cur] copy complete
    __syncthreads();  // everyone done reading et[cur^1] last stage + data visible
    if (stage + 1 < NSTAGES) stage_copy(stage + 1, cur ^ 1);

    #pragma unroll
    for (int kk = 0; kk < CODES_PER_STAGE / 16; ++kk) {
      const int code = (stage << 6) + (kk << 4) + col;
      // B fragment (16-bit B 32x16): lane holds column `col`, channels
      // cs*32 + hi*8 + [0,8) and cs*32 + 16 + hi*8 + [0,8), read as 2x ds_read_b128.
      const unsigned short* erow = &et[cur][(kk << 4) + col][hi * 8];
      v8f acc = {0.f, 0.f, 0.f, 0.f, 0.f, 0.f, 0.f, 0.f};
      #pragma unroll
      for (int cs = 0; cs < 8; ++cs) {
        union { v8bf h[2]; v16bf v; } u;
        u.h[0] = *(const v8bf*)(erow + cs * 32);
        u.h[1] = *(const v8bf*)(erow + cs * 32 + 16);
        acc = __builtin_amdgcn_wmma_f32_16x16x32_bf16(false, afrag[cs], false, u.v,
                                                      (short)0, acc, false, false);
      }
      const float en = enorm[code];
      #pragma unroll
      for (int r = 0; r < 8; ++r) {        // acc slot r -> row m = r + 8*hi
        float d = en + acc[r];             // ||e||^2 - 2 z.e  (||z||^2 dropped)
        if (d < bestd[r]) { bestd[r] = d; besti[r] = code; }
      }
    }
  }

  // Min-reduce across the 16 lanes holding different k columns for same rows;
  // tie-break to lowest code index = jnp.argmin first-occurrence semantics.
  #pragma unroll
  for (int off = 1; off < 16; off <<= 1) {
    #pragma unroll
    for (int r = 0; r < 8; ++r) {
      float od = __shfl_xor(bestd[r], off, 16);
      int   oi = __shfl_xor(besti[r], off, 16);
      if (od < bestd[r] || (od == bestd[r] && oi < besti[r])) {
        bestd[r] = od; besti[r] = oi;
      }
    }
  }
  if (col == 0) {
    const int nbase = bid * 128 + (wave << 4) + (hi << 3);
    #pragma unroll
    for (int r = 0; r < 8; ++r) idx_out[nbase + r] = besti[r];
  }
}

// ---- Kernel 3: z_q gather (fp32 codebook) + per-block loss partials --------
__global__ void __launch_bounds__(256) vq_gather(const float* __restrict__ z,
                                                 const float* __restrict__ emb,
                                                 const int* __restrict__ idx,
                                                 float* __restrict__ zq,
                                                 float* __restrict__ partial) {
  const size_t base = (size_t)blockIdx.x * 1024; // 1024 elems, same (b,c) chunk
  float s = 0.f;
  #pragma unroll
  for (int j = 0; j < 4; ++j) {
    size_t o = base + (size_t)(j << 8) + threadIdx.x;
    int sp = (int)(o & (Ss - 1));
    int c  = (int)((o >> 15) & (Cc - 1));
    int bb = (int)(o >> 23);
    int n  = bb * Ss + sp;
    int k  = idx[n];                 // coalesced along sp
    float e  = emb[k * Cc + c];      // <=256 distinct rows, L2-resident
    float zz = z[o];
    zq[o] = e;                       // straight-through: z_q numerically = emb[idx]
    float d = e - zz;
    s += d * d;
  }
  #pragma unroll
  for (int m = 16; m >= 1; m >>= 1) s += __shfl_xor(s, m, 32);
  __shared__ float red[8];
  if ((threadIdx.x & 31) == 0) red[threadIdx.x >> 5] = s;
  __syncthreads();
  if (threadIdx.x == 0) {
    float t = 0.f;
    #pragma unroll
    for (int i = 0; i < 8; ++i) t += red[i];
    partial[blockIdx.x] = t;
  }
}

// ---- Kernel 4: deterministic fixed-order loss reduction --------------------
__global__ void __launch_bounds__(256) vq_loss_reduce(const float* __restrict__ partial,
                                                      float* __restrict__ loss_out) {
  float s = 0.f;
  for (int i = threadIdx.x; i < ZQ_ELEMS / 1024; i += 256) s += partial[i];
  #pragma unroll
  for (int m = 16; m >= 1; m >>= 1) s += __shfl_xor(s, m, 32);
  __shared__ float red[8];
  if ((threadIdx.x & 31) == 0) red[threadIdx.x >> 5] = s;
  __syncthreads();
  if (threadIdx.x == 0) {
    float t = 0.f;
    for (int i = 0; i < 8; ++i) t += red[i];
    // loss = mean((zq-z)^2) + 0.25*mean((zq-z)^2)
    loss_out[0] = 1.25f * t / (float)ZQ_ELEMS;
  }
}

extern "C" void kernel_launch(void* const* d_in, const int* in_sizes, int n_in,
                              void* d_out, int out_size, void* d_ws, size_t ws_size,
                              hipStream_t stream) {
  const float* z   = (const float*)d_in[0];   // [4,256,32,32,32] fp32
  const float* emb = (const float*)d_in[1];   // [1024,256] fp32

  // Output tuple, flat: z_q (33554432 f32) | loss (1 f32) | idx (131072 i32)
  float* zq   = (float*)d_out;
  float* loss = zq + ZQ_ELEMS;
  int*   idx  = (int*)(zq + ZQ_ELEMS + 1);

  // Workspace: bf16 codebook (512K) | norms (4K) | loss partials (128K)
  unsigned short* ebuf  = (unsigned short*)d_ws;
  float* enorm   = (float*)((char*)d_ws + (size_t)Kk * Cc * 2);
  float* partial = enorm + Kk;

  vq_prep_emb   <<<Kk, 256, 0, stream>>>(emb, ebuf, enorm);
  vq_argmin     <<<Nn / 128, 256, 0, stream>>>(z, ebuf, enorm, idx);
  vq_gather     <<<ZQ_ELEMS / 1024, 256, 0, stream>>>(z, emb, idx, zq, partial);
  vq_loss_reduce<<<1, 256, 0, stream>>>(partial, loss);
}
